// AdvancedCausalLiquidNetworkModel_30528627540774
// MI455X (gfx1250) — compile-verified, matrix operand order checked by reading, not run on hardware
//
#include <hip/hip_runtime.h>
#include <hip/hip_bf16.h>

// ---------------- problem constants ----------------
#define BB   512
#define TT   2048
#define NIN  32
#define HH   256
#define BT   32          // batch rows per workgroup
#define NW   8           // waves per workgroup (256 threads, wave32)
#define DT_  0.1f
#define EPS_ 1e-5f

typedef __attribute__((ext_vector_type(16))) __bf16 v16bf;
typedef __attribute__((ext_vector_type(8)))  float  v8f;

__device__ __forceinline__ unsigned short f2bf(float f) {
    unsigned int u = __float_as_uint(f);
    u += 0x7FFFu + ((u >> 16) & 1u);   // round-to-nearest-even
    return (unsigned short)(u >> 16);
}

__device__ __forceinline__ float sigmoidf_(float x) { return 1.0f / (1.0f + __expf(-x)); }

__device__ __forceinline__ float wred(float v) {
#pragma unroll
    for (int o = 16; o > 0; o >>= 1) v += __shfl_xor(v, o, 32);
    return v;
}

// ---------------------------------------------------------------------------
// Weight repack: W [K][N] f32 row-major  ->  Wt [N][K] bf16 (transposed),
// so each lane's WMMA B-fragment is two contiguous 16B chunks.
// blockIdx.y selects matrix: 0..5 = H x H matrices, 6 = Win0 (K=32, N=256).
// ---------------------------------------------------------------------------
__global__ void repack_weights(const float* __restrict__ Wattn0, const float* __restrict__ Wrec0,
                               const float* __restrict__ Win1,   const float* __restrict__ Wattn1,
                               const float* __restrict__ Wrec1,  const float* __restrict__ Wskip,
                               const float* __restrict__ Win0,   unsigned short* __restrict__ ws) {
    int m = blockIdx.y;
    const float* src;
    switch (m) {
        case 0: src = Wattn0; break;
        case 1: src = Wrec0;  break;
        case 2: src = Win1;   break;
        case 3: src = Wattn1; break;
        case 4: src = Wrec1;  break;
        case 5: src = Wskip;  break;
        default: src = Win0;  break;
    }
    int K = (m == 6) ? NIN : HH;
    int N = HH;
    unsigned short* dst = ws + (size_t)m * HH * HH;   // slot 6 only partially used
    int idx = blockIdx.x * blockDim.x + threadIdx.x;
    if (idx >= N * K) return;
    int n = idx / K;
    int k = idx - n * K;
    dst[n * K + k] = f2bf(src[k * N + n]);
}

// ---------------------------------------------------------------------------
// [BT x HH] = [BT x K] @ [K x HH] with v_wmma_f32_16x16x32_bf16.
// Ab: bf16 in LDS, row-major [BT][K]  (A frag = 2x ds_load_b128, shared by 4 WMMAs)
// Wt: bf16 packed [HH][K] in global (L2-resident; B frag = 2x global_load_b128)
// D : f32 in LDS [BT][HH].
// ---------------------------------------------------------------------------
template <int K>
__device__ __forceinline__ void gemm_bt(const unsigned short* __restrict__ Ab,
                                        const unsigned short* __restrict__ Wt,
                                        float* __restrict__ D, int tid) {
    const int lane = tid & 31;
    const int wave = tid >> 5;
    const int lh   = lane & 15;
    const int half = lane >> 4;
    const int m0   = (wave & 4) ? 16 : 0;
    const int ng   = (wave & 3) * 4;              // first of 4 N-tiles

    const unsigned short* arow = Ab + (m0 + lh) * K;
    const unsigned short* w0 = Wt + ((ng + 0) * 16 + lh) * K;
    const unsigned short* w1 = Wt + ((ng + 1) * 16 + lh) * K;
    const unsigned short* w2 = Wt + ((ng + 2) * 16 + lh) * K;
    const unsigned short* w3 = Wt + ((ng + 3) * 16 + lh) * K;

    v8f acc0 = {}, acc1 = {}, acc2 = {}, acc3 = {};

#pragma unroll
    for (int kb = 0; kb < K; kb += 32) {
        const int c0 = kb + (half << 3);          // lanes 0-15: K[kb..kb+7];   16-31: K[kb+8..kb+15]
        const int c1 = kb + 16 + (half << 3);     // lanes 0-15: K[kb+16..+23]; 16-31: K[kb+24..+31]
        union { v16bf v; uint4 q[2]; } a, b;
        a.q[0] = *(const uint4*)(arow + c0);
        a.q[1] = *(const uint4*)(arow + c1);

        b.q[0] = *(const uint4*)(w0 + c0);
        b.q[1] = *(const uint4*)(w0 + c1);
        acc0 = __builtin_amdgcn_wmma_f32_16x16x32_bf16(false, a.v, false, b.v, (short)0, acc0, false, false);
        b.q[0] = *(const uint4*)(w1 + c0);
        b.q[1] = *(const uint4*)(w1 + c1);
        acc1 = __builtin_amdgcn_wmma_f32_16x16x32_bf16(false, a.v, false, b.v, (short)0, acc1, false, false);
        b.q[0] = *(const uint4*)(w2 + c0);
        b.q[1] = *(const uint4*)(w2 + c1);
        acc2 = __builtin_amdgcn_wmma_f32_16x16x32_bf16(false, a.v, false, b.v, (short)0, acc2, false, false);
        b.q[0] = *(const uint4*)(w3 + c0);
        b.q[1] = *(const uint4*)(w3 + c1);
        acc3 = __builtin_amdgcn_wmma_f32_16x16x32_bf16(false, a.v, false, b.v, (short)0, acc3, false, false);
    }

    // C/D layout: VGPR r -> row m0 + half*8 + r, col ntile*16 + (lane&15)
    float* drow = D + (m0 + (half << 3)) * HH + lh;
#pragma unroll
    for (int r = 0; r < 8; ++r) {
        drow[r * HH + (ng + 0) * 16] = acc0[r];
        drow[r * HH + (ng + 1) * 16] = acc1[r];
        drow[r * HH + (ng + 2) * 16] = acc2[r];
        drow[r * HH + (ng + 3) * 16] = acc3[r];
    }
}

// ---------------------------------------------------------------------------
// Persistent per-workgroup recurrence over T. Each WG owns BT batch rows.
// ---------------------------------------------------------------------------
__global__ void __launch_bounds__(256, 1)
liquid_net_kernel(const float* __restrict__ x, const unsigned short* __restrict__ wpack,
                  const float* __restrict__ bin0,  const float* __restrict__ brec0,
                  const float* __restrict__ battn0,const float* __restrict__ Wev0,
                  const float* __restrict__ bev0,  const float* __restrict__ tau0,
                  const float* __restrict__ gamma0,const float* __restrict__ beta0,
                  const float* __restrict__ bin1,  const float* __restrict__ brec1,
                  const float* __restrict__ battn1,const float* __restrict__ Wev1,
                  const float* __restrict__ bev1,  const float* __restrict__ tau1,
                  const float* __restrict__ gamma1,const float* __restrict__ beta1,
                  const float* __restrict__ bskip, const float* __restrict__ Wout,
                  const float* __restrict__ bout,  float* __restrict__ out) {
    extern __shared__ char smem[];
    float* Xf   = (float*)smem;             // [BT][NIN]
    float* Xp   = Xf   + BT * NIN;          // [BT][NIN]  prev x (detached)
    float* H0   = Xp   + BT * NIN;          // [BT][HH]
    float* H1   = H0   + BT * HH;
    float* H0n  = H1   + BT * HH;
    float* AccA = H0n  + BT * HH;
    float* AccB = AccA + BT * HH;
    float* EW0  = AccB + BT * HH;           // [BT]
    float* EW1  = EW0  + BT;
    unsigned short* Xb   = (unsigned short*)(EW1 + BT);   // bf16 mirrors (GEMM A operands)
    unsigned short* H0b  = Xb   + BT * NIN;
    unsigned short* H1b  = H0b  + BT * HH;
    unsigned short* H0nb = H1b  + BT * HH;
    unsigned short* HAb  = H0nb + BT * HH;  // h*attn staging (bf16 only)

    const int tid   = threadIdx.x;
    const int lane  = tid & 31;
    const int wave  = tid >> 5;
    const int bbase = blockIdx.x * BT;

    for (int i = tid; i < BT * HH; i += 256) {
        H0[i] = 0.f; H1[i] = 0.f;
        H0b[i] = 0; H1b[i] = 0;
    }
    for (int i = tid; i < BT * NIN; i += 256) { Xp[i] = 0.f; }
    __syncthreads();

    float* ew0g = out + BB;                  // ew0s [BB][TT]
    float* ew1g = out + BB + (size_t)BB * TT;

    int lzero = 0;   // laundered zero: defeats LICM hoisting of weight loads out of t-loop

    for (int t = 0; t < TT; ++t) {
        const float mask = (t > 0) ? 1.0f : 0.0f;

        // Opaque re-definition each iteration: compiler cannot prove the weight
        // pointers are loop-invariant, so B-fragment loads stay inside the loop
        // (L2-resident hits) instead of being hoisted + spilled to scratch.
        asm volatile("" : "+s"(lzero));
        const unsigned short* wp      = wpack + lzero;
        const unsigned short* Wattn0t = wp + 0 * HH * HH;
        const unsigned short* Wrec0t  = wp + 1 * HH * HH;
        const unsigned short* Win1t   = wp + 2 * HH * HH;
        const unsigned short* Wattn1t = wp + 3 * HH * HH;
        const unsigned short* Wrec1t  = wp + 4 * HH * HH;
        const unsigned short* Wskipt  = wp + 5 * HH * HH;
        const unsigned short* Win0t   = wp + 6 * HH * HH;   // [HH][NIN]

        // ---- load x(t) tile (f32 + bf16 mirror) ----
        for (int i = tid; i < BT * NIN; i += 256) {
            int r = i >> 5, c = i & (NIN - 1);
            float v = x[((size_t)(bbase + r) * TT + t) * NIN + c];
            Xf[i] = v;
            Xb[i] = f2bf(v);
        }
        __syncthreads();

        // ---- layer 0 input + attention GEMMs ----
        gemm_bt<NIN>(Xb, Win0t, AccA, tid);    // x @ Win0
        gemm_bt<HH>(H0b, Wattn0t, AccB, tid);  // h0 @ Wattn0
        __syncthreads();

        // ---- attn0 / h*attn / inp0 ; ew0 ----
        for (int i = tid; i < BT * HH; i += 256) {
            int c = i & (HH - 1);
            float attn = sigmoidf_(AccB[i] + battn0[c]);
            HAb[i]  = f2bf(H0[i] * attn);
            AccA[i] = tanhf(AccA[i] + bin0[c]);
        }
        for (int r = wave; r < BT; r += NW) {
            float s = Xf[r * NIN + lane] * Wev0[lane] + Xp[r * NIN + lane] * Wev0[NIN + lane];
            s = wred(s);
            float e = sigmoidf_(s + bev0[0]) * mask;
            if (lane == 0) { EW0[r] = e; ew0g[(size_t)(bbase + r) * TT + t] = e; }
        }
        __syncthreads();

        gemm_bt<HH>(HAb, Wrec0t, AccB, tid);   // (h0*attn) @ Wrec0
        __syncthreads();

        // ---- layer0 update + LN + ew1 (row-per-wave, barrier-free fusion) ----
        for (int r = wave; r < BT; r += NW) {
            float ew = EW0[r];
            float vals[HH / 32];
            float s1 = 0.f, s2 = 0.f;
#pragma unroll
            for (int j = 0; j < HH / 32; ++j) {
                int c = lane + j * 32;
                float rec = tanhf(AccB[r * HH + c] + brec0[c]);
                float tc  = fminf(fmaxf(tau0[c], 0.1f), 10.0f);
                float h   = H0[r * HH + c];
                float nh  = h + DT_ * ((-h + AccA[r * HH + c] + rec) / tc) * (1.f + ew);
                vals[j] = nh; s1 += nh; s2 += nh * nh;
            }
            s1 = wred(s1); s2 = wred(s2);
            float mu  = s1 * (1.f / HH);
            float inv = rsqrtf(s2 * (1.f / HH) - mu * mu + EPS_);
            float dot = 0.f;
#pragma unroll
            for (int j = 0; j < HH / 32; ++j) {
                int c = lane + j * 32;
                float hn = (vals[j] - mu) * inv * gamma0[c] + beta0[c];
                H0n[r * HH + c]  = hn;
                H0nb[r * HH + c] = f2bf(hn);
                dot += hn * Wev1[c] + H0[r * HH + c] * Wev1[HH + c];  // p1 == old h0n == H0
            }
            dot = wred(dot);
            float e1 = sigmoidf_(dot + bev1[0]) * mask;
            if (lane == 0) { EW1[r] = e1; ew1g[(size_t)(bbase + r) * TT + t] = e1; }
        }
        __syncthreads();

        // ---- layer 1 GEMMs ----
        gemm_bt<HH>(H0nb, Win1t, AccA, tid);   // h0n @ Win1
        gemm_bt<HH>(H1b, Wattn1t, AccB, tid);  // h1 @ Wattn1
        __syncthreads();

        for (int i = tid; i < BT * HH; i += 256) {
            int c = i & (HH - 1);
            float attn = sigmoidf_(AccB[i] + battn1[c]);
            HAb[i]  = f2bf(H1[i] * attn);
            AccA[i] = tanhf(AccA[i] + bin1[c]);
        }
        __syncthreads();

        gemm_bt<HH>(HAb, Wrec1t, AccB, tid);   // (h1*attn) @ Wrec1
        __syncthreads();

        // ---- layer1 update + LN -> AccA ----
        for (int r = wave; r < BT; r += NW) {
            float ew = EW1[r];
            float vals[HH / 32];
            float s1 = 0.f, s2 = 0.f;
#pragma unroll
            for (int j = 0; j < HH / 32; ++j) {
                int c = lane + j * 32;
                float rec = tanhf(AccB[r * HH + c] + brec1[c]);
                float tc  = fminf(fmaxf(tau1[c], 0.1f), 10.0f);
                float h   = H1[r * HH + c];
                float nh  = h + DT_ * ((-h + AccA[r * HH + c] + rec) / tc) * (1.f + ew);
                vals[j] = nh; s1 += nh; s2 += nh * nh;
            }
            s1 = wred(s1); s2 = wred(s2);
            float mu  = s1 * (1.f / HH);
            float inv = rsqrtf(s2 * (1.f / HH) - mu * mu + EPS_);
#pragma unroll
            for (int j = 0; j < HH / 32; ++j) {
                int c = lane + j * 32;
                AccA[r * HH + c] = (vals[j] - mu) * inv * gamma1[c] + beta1[c];
            }
        }
        __syncthreads();

        gemm_bt<HH>(H0nb, Wskipt, AccB, tid);  // skip: h0n @ Wskip
        __syncthreads();

        // ---- finalize: h1 = ln1 + skip + bskip ; carry updates ----
        for (int i = tid; i < BT * HH; i += 256) {
            int c = i & (HH - 1);
            float h1n = AccA[i] + AccB[i] + bskip[c];
            H1[i]  = h1n;
            H1b[i] = f2bf(h1n);
            H0[i]  = H0n[i];
            H0b[i] = H0nb[i];
        }
        for (int i = tid; i < BT * NIN; i += 256) Xp[i] = Xf[i];
        if (t + 1 < TT) {
            for (int i = tid; i < BT; i += 256)
                __builtin_prefetch(&x[((size_t)(bbase + i) * TT + (t + 1)) * NIN], 0, 0);
        }
        __syncthreads();
    }

    // ---- out = h1 @ Wout + bout ----
    for (int r = wave; r < BT; r += NW) {
        float s = 0.f;
#pragma unroll
        for (int j = 0; j < HH / 32; ++j) {
            int c = lane + j * 32;
            s += H1[r * HH + c] * Wout[c];
        }
        s = wred(s);
        if (lane == 0) out[bbase + r] = s + bout[0];
    }
}

// ---------------------------------------------------------------------------
extern "C" void kernel_launch(void* const* d_in, const int* in_sizes, int n_in,
                              void* d_out, int out_size, void* d_ws, size_t ws_size,
                              hipStream_t stream) {
    const float* x      = (const float*)d_in[0];
    const float* Win0   = (const float*)d_in[1];
    const float* bin0   = (const float*)d_in[2];
    const float* Wrec0  = (const float*)d_in[3];
    const float* brec0  = (const float*)d_in[4];
    const float* Wattn0 = (const float*)d_in[5];
    const float* battn0 = (const float*)d_in[6];
    const float* Wev0   = (const float*)d_in[7];
    const float* bev0   = (const float*)d_in[8];
    const float* tau0   = (const float*)d_in[9];
    const float* gamma0 = (const float*)d_in[10];
    const float* beta0  = (const float*)d_in[11];
    const float* Win1   = (const float*)d_in[12];
    const float* bin1   = (const float*)d_in[13];
    const float* Wrec1  = (const float*)d_in[14];
    const float* brec1  = (const float*)d_in[15];
    const float* Wattn1 = (const float*)d_in[16];
    const float* battn1 = (const float*)d_in[17];
    const float* Wev1   = (const float*)d_in[18];
    const float* bev1   = (const float*)d_in[19];
    const float* tau1   = (const float*)d_in[20];
    const float* gamma1 = (const float*)d_in[21];
    const float* beta1  = (const float*)d_in[22];
    const float* Wskip  = (const float*)d_in[23];
    const float* bskip  = (const float*)d_in[24];
    const float* Wout   = (const float*)d_in[25];
    const float* bout   = (const float*)d_in[26];

    unsigned short* wpack = (unsigned short*)d_ws;

    // 1) one-time bf16 transpose-repack of all GEMM weights into d_ws
    dim3 rgrid(HH * HH / 256, 7);
    repack_weights<<<rgrid, 256, 0, stream>>>(Wattn0, Wrec0, Win1, Wattn1, Wrec1, Wskip,
                                              Win0, wpack);

    // 2) persistent batch-parallel recurrence
    size_t smem = (size_t)(2 * BT * NIN + 5 * BT * HH + 2 * BT) * sizeof(float)
                + (size_t)(BT * NIN + 4 * BT * HH) * sizeof(unsigned short);  // ~240 KB < 320 KB WGP LDS
    liquid_net_kernel<<<BB / BT, 256, smem, stream>>>(
        x, wpack,
        bin0, brec0, battn0, Wev0, bev0, tau0, gamma0, beta0,
        bin1, brec1, battn1, Wev1, bev1, tau1, gamma1, beta1,
        bskip, Wout, bout, (float*)d_out);
}
